// MAB_8005819040034
// MI455X (gfx1250) — compile-verified, matrix operand order checked by reading, not run on hardware
//
#include <hip/hip_runtime.h>
#include <hip/hip_bf16.h>

typedef __attribute__((ext_vector_type(16))) _Float16 v16h;
typedef __attribute__((ext_vector_type(8)))  _Float16 v8h;
typedef __attribute__((ext_vector_type(4)))  _Float16 v4h;
typedef __attribute__((ext_vector_type(8)))  float    v8f;

#define BB    2
#define NQS   4096
#define NKS   4096
#define DVS   256
#define NH    8
#define DHH   32
#define LNEPS 1e-5f
#define QSCALE 0.17677669529663687f   // 1/sqrt(32)

// D = A*B + C, f16 inputs, f32 accumulate (v_wmma_f32_16x16x32_f16)
static __device__ __forceinline__ v8f wmma16(v16h a, v16h b, v8f c) {
  return __builtin_amdgcn_wmma_f32_16x16x32_f16(false, a, false, b, (short)0, c,
                                                false, false);
}

// A/B operand loader, f16 source, ISA 16-bit 16x32 striping:
// hi = lane/16, k0 = 8*hi; halves[0..7]=row[kb+k0+i], halves[8..15]=row[kb+k0+16+i]
static __device__ __forceinline__ v16h load_ab_f16(const _Float16* row, int kb, int hi) {
  v8h lo = *(const v8h*)(row + kb + 8 * hi);
  v8h hh = *(const v8h*)(row + kb + 8 * hi + 16);
  v16h a;
#pragma unroll
  for (int i = 0; i < 8; ++i) { a[i] = lo[i]; a[8 + i] = hh[i]; }
  return a;
}

static __device__ __forceinline__ float redmax16(float v) {
#pragma unroll
  for (int m = 1; m < 16; m <<= 1) v = fmaxf(v, __shfl_xor(v, m, 32));
  return v;
}
static __device__ __forceinline__ float redsum16(float v) {
#pragma unroll
  for (int m = 1; m < 16; m <<= 1) v += __shfl_xor(v, m, 32);
  return v;
}

// ---------------------------------------------------------------------------
// Kernel 0: f32 -> f16 conversion (optionally scaled).  Grid-stride, float4.
// ---------------------------------------------------------------------------
__global__ __launch_bounds__(256) void mab_cvt_kernel(const float* __restrict__ src,
                                                      _Float16* __restrict__ dst,
                                                      int n4, float scale) {
  int i = blockIdx.x * blockDim.x + threadIdx.x;
  const int stride = gridDim.x * blockDim.x;
  const float4* s4 = (const float4*)src;
  v4h* d4 = (v4h*)dst;
  for (; i < n4; i += stride) {
    float4 v = s4[i];
    v4h o;
    o[0] = (_Float16)(v.x * scale);
    o[1] = (_Float16)(v.y * scale);
    o[2] = (_Float16)(v.z * scale);
    o[3] = (_Float16)(v.w * scale);
    d4[i] = o;
  }
}

// ---------------------------------------------------------------------------
// Kernel 1: Q/K/V projections from pre-converted f16.  Wave = 32x64 tile.
// 3072 waves = 3 proj * 256 rowtiles * 4 colgroups; 384 blocks of 8 waves.
// ---------------------------------------------------------------------------
__global__ __launch_bounds__(256) void mab_proj_kernel(
    const _Float16* __restrict__ Xq, const _Float16* __restrict__ Xk,
    const _Float16* __restrict__ Wq16, const _Float16* __restrict__ Wk16,
    const _Float16* __restrict__ Wv16,
    const float* __restrict__ bq, const float* __restrict__ bk,
    const float* __restrict__ bv,
    _Float16* __restrict__ Qh, _Float16* __restrict__ Kh,
    _Float16* __restrict__ Vt) {
  const int lane = threadIdx.x & 31;
  const int wid  = threadIdx.x >> 5;
  const int w    = blockIdx.x * 8 + wid;
  const int proj = w >> 10;          // 0=Q,1=K,2=V  (1024 waves each)
  const int rem  = w & 1023;
  const int rowtile = rem >> 2;      // 0..255 (32 rows each over B*N=8192)
  const int cg   = rem & 3;          // 64-col group
  const int hi   = lane >> 4;
  const int ln   = lane & 15;

  const _Float16* X = (proj == 0) ? Xq : Xk;
  const _Float16* W = (proj == 0) ? Wq16 : (proj == 1) ? Wk16 : Wv16;
  const float* bias = (proj == 0) ? bq : (proj == 1) ? bk : bv;
  const float bscale = (proj == 0) ? QSCALE : 1.0f;  // Wq16 is pre-scaled

  const int rowbase = rowtile * 32;
  const _Float16* arow0 = X + (size_t)(rowbase + ln) * DVS;
  const _Float16* arow1 = X + (size_t)(rowbase + 16 + ln) * DVS;

  v8f acc[2][4];
#pragma unroll
  for (int i = 0; i < 2; ++i)
#pragma unroll
    for (int nt = 0; nt < 4; ++nt) acc[i][nt] = (v8f){};

  for (int ks = 0; ks < DVS; ks += 32) {
    v16h a0 = load_ab_f16(arow0, ks, hi);
    v16h a1 = load_ab_f16(arow1, ks, hi);
#pragma unroll
    for (int nt = 0; nt < 4; ++nt) {
      const int c = cg * 64 + nt * 16 + ln;          // output feature (B column)
      v16h b = load_ab_f16(W + (size_t)c * DVS, ks, hi);
      acc[0][nt] = wmma16(a0, b, acc[0][nt]);
      acc[1][nt] = wmma16(a1, b, acc[1][nt]);
    }
  }

#pragma unroll
  for (int nt = 0; nt < 4; ++nt) {
    const int c = cg * 64 + nt * 16 + ln;
    const int h = c >> 5, d = c & 31;
    const float bia = bias[c] * bscale;
#pragma unroll
    for (int i = 0; i < 2; ++i) {
#pragma unroll
      for (int r = 0; r < 8; ++r) {
        const int row = rowbase + i * 16 + r + 8 * hi;  // flat token over B*N
        const int b_  = row >> 12;
        const int n   = row & 4095;
        const float v = acc[i][nt][r] + bia;
        if (proj == 2) {
          Vt[(((size_t)(b_ * NH + h)) * DHH + d) * NKS + n] = (_Float16)v;
        } else {
          _Float16* dst = (proj == 0) ? Qh : Kh;
          dst[(((size_t)(b_ * NH + h)) * NQS + n) * DHH + d] = (_Float16)v;
        }
      }
    }
  }
}

// ---------------------------------------------------------------------------
// Kernel 2: flash attention.  Wave = (b,h, 16-query tile).  Two 32-key
// chunks per iteration with independent LDS staging buffers.
// 4096 waves; 512 blocks of 8 waves.
// ---------------------------------------------------------------------------
struct AttnState {
  v8f O0, O1;
  float m[8], l[8];
};

static __device__ __forceinline__ void attn_chunk(
    const v16h& aq, const _Float16* Kb, const _Float16* Vb, int kb,
    _Float16* pl, int hi, int ln, AttnState& st) {
  // S = Q * K^T : B columns are key rows of Kh (contiguous halves)
  v16h b0 = load_ab_f16(Kb + (size_t)(kb + ln) * DHH, 0, hi);
  v16h b1 = load_ab_f16(Kb + (size_t)(kb + 16 + ln) * DHH, 0, hi);
  v8f z  = (v8f){};
  v8f s0 = wmma16(aq, b0, z);
  v8f s1 = wmma16(aq, b1, z);

  float p0[8], p1[8];
#pragma unroll
  for (int r = 0; r < 8; ++r) {
    const float v0 = s0[r], v1 = s1[r];               // Q pre-scaled by 1/sqrt(dh)
    float mx = redmax16(fmaxf(v0, v1));
    const float mn    = fmaxf(st.m[r], mx);
    const float alpha = __expf(st.m[r] - mn);
    const float e0 = __expf(v0 - mn), e1 = __expf(v1 - mn);
    const float rs = redsum16(e0 + e1);
    st.l[r] = st.l[r] * alpha + rs;
    st.m[r] = mn;
    st.O0[r] *= alpha; st.O1[r] *= alpha;
    p0[r] = e0; p1[r] = e1;
  }

  // stage P (C-layout coords) into LDS as row-major 16x32, re-read A-layout
#pragma unroll
  for (int r = 0; r < 8; ++r) {
    const int row = r + 8 * hi;
    pl[row * 32 + ln]      = (_Float16)p0[r];
    pl[row * 32 + 16 + ln] = (_Float16)p1[r];
  }
  v16h ap = load_ab_f16(pl + ln * 32, 0, hi);

  // O += P * V : B columns are dh rows of Vt (contiguous over keys)
  v16h bv0 = load_ab_f16(Vb + (size_t)ln * NKS, kb, hi);
  v16h bv1 = load_ab_f16(Vb + (size_t)(ln + 16) * NKS, kb, hi);
  st.O0 = wmma16(ap, bv0, st.O0);
  st.O1 = wmma16(ap, bv1, st.O1);
}

__global__ __launch_bounds__(256) void mab_attn_kernel(
    const _Float16* __restrict__ Qh, const _Float16* __restrict__ Kh,
    const _Float16* __restrict__ Vt, _Float16* __restrict__ Oh) {
  __shared__ __align__(16) _Float16 lds[8 * 1024];   // 2 x 1KB per wave
  const int lane  = threadIdx.x & 31;
  const int wid   = threadIdx.x >> 5;
  const int w     = blockIdx.x * 8 + wid;
  const int bh    = w >> 8;                          // 0..15
  const int qtile = w & 255;
  const int hi    = lane >> 4;
  const int ln    = lane & 15;
  _Float16* pl0 = lds + wid * 1024;
  _Float16* pl1 = pl0 + 512;

  const _Float16* Qb = Qh + ((size_t)bh * NQS + qtile * 16 + ln) * DHH;
  const _Float16* Kb = Kh + (size_t)bh * NKS * DHH;
  const _Float16* Vb = Vt + (size_t)bh * DHH * NKS;

  const v16h aq = load_ab_f16(Qb, 0, hi);            // Q tile, persistent

  AttnState st;
  st.O0 = (v8f){}; st.O1 = (v8f){};
#pragma unroll
  for (int r = 0; r < 8; ++r) { st.m[r] = -1e30f; st.l[r] = 0.0f; }

  for (int kb = 0; kb < NKS; kb += 64) {
    attn_chunk(aq, Kb, Vb, kb,      pl0, hi, ln, st);
    attn_chunk(aq, Kb, Vb, kb + 32, pl1, hi, ln, st);
  }

  const int b_ = bh >> 3, h = bh & 7;
#pragma unroll
  for (int r = 0; r < 8; ++r) {
    const int row   = qtile * 16 + r + 8 * hi;
    const float inv = 1.0f / st.l[r];
    const size_t base = ((size_t)b_ * NQS + row) * DVS + h * DHH;
    Oh[base + ln]      = (_Float16)(st.O0[r] * inv);
    Oh[base + 16 + ln] = (_Float16)(st.O1[r] * inv);
  }
}

// ---------------------------------------------------------------------------
// Kernel 3: out = LN1( y + relu(y) ),  y = LN0( O @ Wo^T + bo ).
// Wave owns a full 16x256 row tile; LayerNorms reduce in-register.
// 512 waves; 64 blocks.
// ---------------------------------------------------------------------------
__global__ __launch_bounds__(256) void mab_out_kernel(
    const _Float16* __restrict__ Oh, const _Float16* __restrict__ Wo16,
    const float* __restrict__ bo, const float* __restrict__ g0,
    const float* __restrict__ beta0, const float* __restrict__ g1,
    const float* __restrict__ beta1, float* __restrict__ out) {
  const int lane = threadIdx.x & 31;
  const int wid  = threadIdx.x >> 5;
  const int rowtile = blockIdx.x * 8 + wid;          // 0..511
  const int hi = lane >> 4, ln = lane & 15;

  const _Float16* arow = Oh + (size_t)(rowtile * 16 + ln) * DVS;

  v8f acc[16];
#pragma unroll
  for (int nt = 0; nt < 16; ++nt) acc[nt] = (v8f){};

  for (int ks = 0; ks < DVS; ks += 32) {
    v16h a = load_ab_f16(arow, ks, hi);
#pragma unroll
    for (int nt = 0; nt < 16; ++nt) {
      const int c = nt * 16 + ln;
      v16h b = load_ab_f16(Wo16 + (size_t)c * DVS, ks, hi);
      acc[nt] = wmma16(a, b, acc[nt]);
    }
  }

#pragma unroll
  for (int nt = 0; nt < 16; ++nt) {
    const float bia = bo[nt * 16 + ln];
#pragma unroll
    for (int r = 0; r < 8; ++r) acc[nt][r] += bia;
  }

#pragma unroll
  for (int r = 0; r < 8; ++r) {
    // LN0
    float s = 0.0f;
#pragma unroll
    for (int nt = 0; nt < 16; ++nt) s += acc[nt][r];
    const float mu = redsum16(s) * (1.0f / 256.0f);
    float vs = 0.0f;
#pragma unroll
    for (int nt = 0; nt < 16; ++nt) { float d = acc[nt][r] - mu; vs += d * d; }
    const float rstd = rsqrtf(redsum16(vs) * (1.0f / 256.0f) + LNEPS);
#pragma unroll
    for (int nt = 0; nt < 16; ++nt) {
      const int c = nt * 16 + ln;
      const float y = (acc[nt][r] - mu) * rstd * g0[c] + beta0[c];
      acc[nt][r] = y + fmaxf(y, 0.0f);               // y + relu(y)
    }
    // LN1
    float s2 = 0.0f;
#pragma unroll
    for (int nt = 0; nt < 16; ++nt) s2 += acc[nt][r];
    const float mu2 = redsum16(s2) * (1.0f / 256.0f);
    float vs2 = 0.0f;
#pragma unroll
    for (int nt = 0; nt < 16; ++nt) { float d = acc[nt][r] - mu2; vs2 += d * d; }
    const float rstd2 = rsqrtf(redsum16(vs2) * (1.0f / 256.0f) + LNEPS);
    const int row = rowtile * 16 + r + 8 * hi;
#pragma unroll
    for (int nt = 0; nt < 16; ++nt) {
      const int c = nt * 16 + ln;
      out[(size_t)row * DVS + c] = (acc[nt][r] - mu2) * rstd2 * g1[c] + beta1[c];
    }
  }
}

// ---------------------------------------------------------------------------
extern "C" void kernel_launch(void* const* d_in, const int* in_sizes, int n_in,
                              void* d_out, int out_size, void* d_ws, size_t ws_size,
                              hipStream_t stream) {
  const float* Q     = (const float*)d_in[0];
  const float* K     = (const float*)d_in[1];
  const float* Wq    = (const float*)d_in[2];
  const float* bq    = (const float*)d_in[3];
  const float* Wk    = (const float*)d_in[4];
  const float* bk    = (const float*)d_in[5];
  const float* Wv    = (const float*)d_in[6];
  const float* bv    = (const float*)d_in[7];
  const float* Wo    = (const float*)d_in[8];
  const float* bo    = (const float*)d_in[9];
  const float* g0    = (const float*)d_in[10];
  const float* beta0 = (const float*)d_in[11];
  const float* g1    = (const float*)d_in[12];
  const float* beta1 = (const float*)d_in[13];

  // workspace carve-out (f16):
  //   Xq | Xk : 2M halves each   (inputs, converted)
  //   Wq16|Wk16|Wv16|Wo16 : 64K halves each (weights, Wq pre-scaled)
  //   Qh | Kh | Vt | Oh : 2M halves each
  const size_t nTok = (size_t)BB * NQS * DVS;        // 2,097,152
  const size_t nW   = (size_t)DVS * DVS;             // 65,536
  _Float16* Xq16 = (_Float16*)d_ws;
  _Float16* Xk16 = Xq16 + nTok;
  _Float16* Wq16 = Xk16 + nTok;
  _Float16* Wk16 = Wq16 + nW;
  _Float16* Wv16 = Wk16 + nW;
  _Float16* Wo16 = Wv16 + nW;
  _Float16* Qh   = Wo16 + nW;
  _Float16* Kh   = Qh + nTok;
  _Float16* Vt   = Kh + nTok;
  _Float16* Oh   = Vt + nTok;

  // pre-convert everything to f16 (memory-bound, ~25MB total)
  mab_cvt_kernel<<<512, 256, 0, stream>>>(Q,  Xq16, (int)(nTok / 4), 1.0f);
  mab_cvt_kernel<<<512, 256, 0, stream>>>(K,  Xk16, (int)(nTok / 4), 1.0f);
  mab_cvt_kernel<<<64, 256, 0, stream>>>(Wq, Wq16, (int)(nW / 4), QSCALE);
  mab_cvt_kernel<<<64, 256, 0, stream>>>(Wk, Wk16, (int)(nW / 4), 1.0f);
  mab_cvt_kernel<<<64, 256, 0, stream>>>(Wv, Wv16, (int)(nW / 4), 1.0f);
  mab_cvt_kernel<<<64, 256, 0, stream>>>(Wo, Wo16, (int)(nW / 4), 1.0f);

  mab_proj_kernel<<<384, 256, 0, stream>>>(Xq16, Xk16, Wq16, Wk16, Wv16,
                                           bq, bk, bv, Qh, Kh, Vt);
  mab_attn_kernel<<<512, 256, 0, stream>>>(Qh, Kh, Vt, Oh);
  mab_out_kernel<<<64, 256, 0, stream>>>(Oh, Wo16, bo, g0, beta0, g1, beta1,
                                         (float*)d_out);
}
